// RNN_50955491999949
// MI455X (gfx1250) — matrix operand order repaired and checked by reading, not compiled, not run on hardware
//
#include <hip/hip_runtime.h>
#include <hip/hip_bf16.h>
#include <math.h>

// Problem constants
#define B_ 64
#define T_ 1024
#define I_ 128
#define H_ 512
#define O_ 64
#define NWG_SCAN 8

typedef __attribute__((ext_vector_type(16))) __bf16 v16bf;
typedef __attribute__((ext_vector_type(8)))  __bf16 v8bf;
typedef __attribute__((ext_vector_type(8)))  float  v8f;

__device__ __forceinline__ v16bf concat8(v8bf lo, v8bf hi) {
    return __builtin_shufflevector(lo, hi, 0,1,2,3,4,5,6,7,8,9,10,11,12,13,14,15);
}

__device__ __forceinline__ v8f wmma_bf16(v16bf a, v16bf b, v8f c) {
    return __builtin_amdgcn_wmma_f32_16x16x32_bf16(false, a, false, b, (short)0, c, false, false);
}

// gfx1250 async copy global->LDS (16B), tracked by ASYNCcnt.
// VDST VGPR carries the LDS byte offset (low 32 bits of a generic LDS pointer).
__device__ __forceinline__ void async_copy_b128(void* ldsdst, const void* gsrc) {
    unsigned loff = (unsigned)(uintptr_t)ldsdst;
    unsigned long long ga = (unsigned long long)(uintptr_t)gsrc;
    asm volatile("global_load_async_to_lds_b128 %0, %1, off"
                 :: "v"(loff), "v"(ga) : "memory");
}
__device__ __forceinline__ void wait_async0() {
    asm volatile("s_wait_asynccnt 0" ::: "memory");
}

// Release/acquire grid barrier over a monotonically increasing counter.
__device__ __forceinline__ void grid_barrier(int* cnt, int target) {
    __syncthreads();
    if (threadIdx.x == 0) {
        __hip_atomic_fetch_add(cnt, 1, __ATOMIC_RELEASE, __HIP_MEMORY_SCOPE_AGENT);
        while (__hip_atomic_load(cnt, __ATOMIC_ACQUIRE, __HIP_MEMORY_SCOPE_AGENT) < target)
            __builtin_amdgcn_s_sleep(1);
    }
    __syncthreads();
}

// f32 -> bf16 weight conversion
__global__ void f32_to_bf16(const float* __restrict__ s, __bf16* __restrict__ d, int n) {
    int i = blockIdx.x * 256 + threadIdx.x;
    if (i < n) d[i] = (__bf16)s[i];
}

// xp[t,b,h] = sum_k in[t,b,k] * W[h,k] + bi[h] + bh[h]
// One workgroup per timestep t (64 rows). Input staged to LDS as bf16 so the
// kernel is safe in-place (in == out for layers 1/2). Output row-major (T*B, H).
// Each wave holds 8 f32 accumulator tiles so one A fragment feeds 8 WMMAs.
__global__ __launch_bounds__(256) void xp_gemm(const float* in, long strideT, long strideB, int K,
                                               const __bf16* __restrict__ Wbf,
                                               const float* __restrict__ bi, const float* __restrict__ bh,
                                               float* out) {
    __shared__ __bf16 smem[B_ * H_];          // up to 64x512 bf16 = 64KB
    const int t   = blockIdx.x;
    const int tid = threadIdx.x;

    const long base = (long)t * strideT;
    for (int idx = tid; idx < B_ * K; idx += 256) {
        int row = idx / K, k = idx - row * K;
        smem[idx] = (__bf16)in[base + (long)row * strideB + k];
    }
    __syncthreads();

    const int w   = tid >> 5;                 // wave 0..7
    const int l   = tid & 31;
    const int m   = w & 3;                    // 16-row batch tile
    const int grp = w >> 2;                   // 0/1 -> n-tiles [0,16) or [16,32)
    const int lr  = l & 15;
    const int hi  = l >> 4;
    const int arow = m * 16 + lr;
    const int r0  = t * B_;
    const int ktiles = K >> 5;

    const __bf16* aptr = smem + arow * K + hi * 16;

    for (int half = 0; half < 2; ++half) {
        const int j0 = grp * 16 + half * 8;
        v8f acc[8] = {{}, {}, {}, {}, {}, {}, {}, {}};
        for (int kt = 0; kt < ktiles; ++kt) {
            int kb = kt * 32;
            v8bf alo = *(const v8bf*)(aptr + kb);          // K = kb + hi*16 + e
            v8bf ahi = *(const v8bf*)(aptr + kb + 8);      // K = kb + hi*16 + 8 + e
            v16bf a  = concat8(alo, ahi);
            #pragma unroll
            for (int jj = 0; jj < 8; ++jj) {
                const __bf16* wrow = Wbf + (long)((j0 + jj) * 16 + lr) * K + hi * 16;
                v16bf b = *(const v16bf*)(wrow + kb);      // K = kb + hi*16 + e
                acc[jj] = wmma_bf16(a, b, acc[jj]);
            }
        }
        #pragma unroll
        for (int jj = 0; jj < 8; ++jj) {
            int col = (j0 + jj) * 16 + lr;
            float bias = bi[col] + bh[col];
            #pragma unroll
            for (int v = 0; v < 8; ++v) {
                int r = r0 + m * 16 + v + 8 * hi;          // D: M = v + 8*hi, N = lr
                out[(long)r * H_ + col] = acc[jj][v] + bias;
            }
        }
    }
}

// Persistent recurrent scan: h = tanh(xp[t] + h @ Wh^T), 1024 dependent steps.
// 8 workgroups, each owns a 64-col slice of Wh (bf16, async-copied into LDS and
// resident for the whole scan). Hidden state double-buffered in global bf16;
// per-step sync via atomic barrier. seq holds xp on entry; overwritten with ys.
__global__ __launch_bounds__(256) void rnn_scan(float* seq, const __bf16* __restrict__ Whbf,
                                                __bf16* hbf, const float* __restrict__ h0l,
                                                int* counter, int T) {
    __shared__ __bf16 wsl[64 * H_];           // 64 cols x 512 k = 64KB
    const int g   = blockIdx.x;
    const int tid = threadIdx.x;

    // async copy Wh slice (rows g*64 .. g*64+63) into LDS: 4096 x 16B chunks
    {
        const __bf16* src = Whbf + (long)g * 64 * H_;
        for (int i = tid; i < (64 * H_) / 8; i += 256)
            async_copy_b128(&wsl[i * 8], src + i * 8);
    }

    // init h buffer 0 (bf16) for our column slice while the copy is in flight
    for (int idx = tid; idx < B_ * 64; idx += 256) {
        int b = idx >> 6, c = idx & 63;
        int col = g * 64 + c;
        hbf[b * H_ + col] = (__bf16)h0l[b * H_ + col];
    }
    wait_async0();
    grid_barrier(counter, NWG_SCAN);

    const int w   = tid >> 5;
    const int l   = tid & 31;
    const int m   = w & 3;
    const int grp = w >> 2;
    const int lr  = l & 15;
    const int hi  = l >> 4;
    const int brow = m * 16 + lr;

    for (int t = 0; t < T; ++t) {
        const __bf16* hc = hbf + (t & 1) * (B_ * H_);
        __bf16*       hn = hbf + ((t + 1) & 1) * (B_ * H_);
        v8f acc[2] = {{}, {}};
        const __bf16* aptr = hc + brow * H_ + hi * 16;
        const __bf16* b0 = wsl + ((grp * 2 + 0) * 16 + lr) * H_ + hi * 16;
        const __bf16* b1 = wsl + ((grp * 2 + 1) * 16 + lr) * H_ + hi * 16;
        for (int kt = 0; kt < H_ / 32; ++kt) {
            int kb = kt * 32;
            if (kt + 1 < H_ / 32) __builtin_prefetch(aptr + kb + 32, 0, 1);
            v8bf alo = *(const v8bf*)(aptr + kb);
            v8bf ahi = *(const v8bf*)(aptr + kb + 8);
            v16bf a  = concat8(alo, ahi);
            v16bf bb0 = *(const v16bf*)(b0 + kb);
            v16bf bb1 = *(const v16bf*)(b1 + kb);
            acc[0] = wmma_bf16(a, bb0, acc[0]);
            acc[1] = wmma_bf16(a, bb1, acc[1]);
        }
        long rowbase = ((long)t * B_) * H_;
        #pragma unroll
        for (int j = 0; j < 2; ++j) {
            int col = g * 64 + (grp * 2 + j) * 16 + lr;
            #pragma unroll
            for (int v = 0; v < 8; ++v) {
                int b = m * 16 + v + 8 * hi;
                float x = acc[j][v] + seq[rowbase + (long)b * H_ + col];
                float y = tanhf(x);
                seq[rowbase + (long)b * H_ + col] = y;   // ys[t] overwrites xp[t]
                hn[b * H_ + col] = (__bf16)y;
            }
        }
        grid_barrier(counter, NWG_SCAN * (t + 2));
    }
}

// out[b,o] = sigmoid(last[b,:] . wfc[o,:] + bfc[o]); 64x64, trivial
__global__ __launch_bounds__(256) void fc_sigmoid(const float* __restrict__ last,
                                                  const float* __restrict__ wfc,
                                                  const float* __restrict__ bfc,
                                                  float* __restrict__ outp) {
    int idx = blockIdx.x * 256 + threadIdx.x;  // 0..4095
    int b = idx >> 6, o = idx & 63;
    float s = bfc[o];
    const float* lrow = last + b * H_;
    const float* wrow = wfc + o * H_;
    for (int k = 0; k < H_; ++k) s += lrow[k] * wrow[k];
    outp[idx] = 1.0f / (1.0f + expf(-s));
}

extern "C" void kernel_launch(void* const* d_in, const int* in_sizes, int n_in,
                              void* d_out, int out_size, void* d_ws, size_t ws_size,
                              hipStream_t stream) {
    const float* x   = (const float*)d_in[0];
    const float* h0  = (const float*)d_in[1];
    const float* wih[3] = {(const float*)d_in[2], (const float*)d_in[6],  (const float*)d_in[10]};
    const float* whh[3] = {(const float*)d_in[3], (const float*)d_in[7],  (const float*)d_in[11]};
    const float* bih[3] = {(const float*)d_in[4], (const float*)d_in[8],  (const float*)d_in[12]};
    const float* bhh[3] = {(const float*)d_in[5], (const float*)d_in[9],  (const float*)d_in[13]};
    const float* wfc = (const float*)d_in[14];
    const float* bfc = (const float*)d_in[15];
    float* out = (float*)d_out;

    char* ws = (char*)d_ws;
    size_t off = 0;
    float* bufS = (float*)(ws + off); off += (size_t)T_ * B_ * H_ * sizeof(float);  // 128MB seq/xp
    __bf16* wihbf[3]; __bf16* whhbf[3];
    wihbf[0] = (__bf16*)(ws + off); off += (size_t)H_ * I_ * 2;
    whhbf[0] = (__bf16*)(ws + off); off += (size_t)H_ * H_ * 2;
    wihbf[1] = (__bf16*)(ws + off); off += (size_t)H_ * H_ * 2;
    whhbf[1] = (__bf16*)(ws + off); off += (size_t)H_ * H_ * 2;
    wihbf[2] = (__bf16*)(ws + off); off += (size_t)H_ * H_ * 2;
    whhbf[2] = (__bf16*)(ws + off); off += (size_t)H_ * H_ * 2;
    __bf16* hbf = (__bf16*)(ws + off); off += (size_t)2 * B_ * H_ * 2;
    int* counter = (int*)(ws + off); off += 64;

    // weight conversions (f32 -> bf16)
    f32_to_bf16<<<(H_ * I_ + 255) / 256, 256, 0, stream>>>(wih[0], wihbf[0], H_ * I_);
    f32_to_bf16<<<(H_ * H_ + 255) / 256, 256, 0, stream>>>(whh[0], whhbf[0], H_ * H_);
    f32_to_bf16<<<(H_ * H_ + 255) / 256, 256, 0, stream>>>(wih[1], wihbf[1], H_ * H_);
    f32_to_bf16<<<(H_ * H_ + 255) / 256, 256, 0, stream>>>(whh[1], whhbf[1], H_ * H_);
    f32_to_bf16<<<(H_ * H_ + 255) / 256, 256, 0, stream>>>(wih[2], wihbf[2], H_ * H_);
    f32_to_bf16<<<(H_ * H_ + 255) / 256, 256, 0, stream>>>(whh[2], whhbf[2], H_ * H_);

    for (int l = 0; l < 3; ++l) {
        if (l == 0) {
            // x is (B,T,I): addr = b*(T*I) + t*I + i
            xp_gemm<<<T_, 256, 0, stream>>>(x, (long)I_, (long)T_ * I_, I_,
                                            wihbf[0], bih[0], bhh[0], bufS);
        } else {
            // prev ys in (T,B,H): addr = t*(B*H) + b*H + k ; in-place
            xp_gemm<<<T_, 256, 0, stream>>>(bufS, (long)B_ * H_, (long)H_, H_,
                                            wihbf[l], bih[l], bhh[l], bufS);
        }
        hipMemsetAsync(counter, 0, sizeof(int), stream);
        rnn_scan<<<NWG_SCAN, 256, 0, stream>>>(bufS, whhbf[l], hbf,
                                               h0 + (size_t)l * B_ * H_, counter, T_);
        // hn[l] = h_T = ys[T-1] (contiguous 64x512 block)
        hipMemcpyAsync(out + B_ * O_ + (size_t)l * B_ * H_,
                       bufS + (size_t)(T_ - 1) * B_ * H_,
                       (size_t)B_ * H_ * sizeof(float),
                       hipMemcpyDeviceToDevice, stream);
    }

    fc_sigmoid<<<(B_ * O_) / 256, 256, 0, stream>>>(bufS + (size_t)(T_ - 1) * B_ * H_,
                                                    wfc, bfc, out);
}
